// Inversion_13159779795723
// MI455X (gfx1250) — matrix operand order repaired and checked by reading, not compiled, not run on hardware
//
#include <hip/hip_runtime.h>
#include <hip/hip_bf16.h>
#include <math.h>

typedef _Float16 half8  __attribute__((ext_vector_type(8)));
typedef _Float16 half16 __attribute__((ext_vector_type(16)));
typedef float    f32x8  __attribute__((ext_vector_type(8)));

#define NZ      1000
#define NZP     1024
#define BATCH   2048
#define TRUNC   50
#define CONST_C 50.0f
#define BN_EPS  1e-5f

// ============================================================
// 1) top-k(50) -> log/clamp/shift -> scatter into f16 z [B][1024]
// ============================================================
__global__ __launch_bounds__(256) void topk_kernel(const float* __restrict__ x,
                                                   _Float16* __restrict__ z) {
  __shared__ float sv[NZP];
  __shared__ float rv[256];
  __shared__ int   ri[256];
  __shared__ float topv[TRUNC];
  __shared__ int   topi[TRUNC];
  __shared__ float s_shift;
  const int b = blockIdx.x, t = threadIdx.x;
  for (int i = t; i < NZP; i += 256) {
    sv[i] = (i < NZ) ? x[(size_t)b * NZ + i] : -1e30f;
    z[(size_t)b * NZP + i] = (_Float16)0.0f;
  }
  __syncthreads();
  for (int sel = 0; sel < TRUNC; ++sel) {
    float mv = -1e30f; int mi = 0;
    for (int i = t; i < NZP; i += 256) { float v = sv[i]; if (v > mv) { mv = v; mi = i; } }
    rv[t] = mv; ri[t] = mi; __syncthreads();
    for (int s = 128; s > 0; s >>= 1) {
      if (t < s && rv[t + s] > rv[t]) { rv[t] = rv[t + s]; ri[t] = ri[t + s]; }
      __syncthreads();
    }
    if (t == 0) { topv[sel] = rv[0]; topi[sel] = ri[0]; sv[ri[0]] = -1e30f; }
    __syncthreads();
  }
  if (t < TRUNC) {
    float v = fmaxf(logf(topv[t]), -1000.0f) + CONST_C;
    topv[t] = v;
  }
  __syncthreads();
  if (t == 0) {
    float mn = topv[0];
    for (int i = 1; i < TRUNC; ++i) mn = fminf(mn, topv[i]);
    s_shift = fmaxf(0.0f, -mn);
  }
  __syncthreads();
  if (t < TRUNC) z[(size_t)b * NZP + topi[t]] = (_Float16)(topv[t] + s_shift);
}

// ============================================================
// 2) weight packing (f32 -> f16, WMMA-friendly [slice][K][N])
// ============================================================
// w1 (1000,512,4,4) -> w1p[pix][ci(1024 pad)][co(512)]
__global__ void pack_w1_kernel(const float* __restrict__ w1, _Float16* __restrict__ wp) {
  size_t idx = (size_t)blockIdx.x * 256 + threadIdx.x;
  const size_t total = 16ull * NZP * 512;
  if (idx >= total) return;
  int co = (int)(idx % 512);
  size_t r = idx / 512;
  int ci = (int)(r % NZP);
  int p  = (int)(r / NZP);
  float v = (ci < NZ) ? w1[(size_t)ci * 8192 + (size_t)co * 16 + p] : 0.0f;
  wp[idx] = (_Float16)v;
}
// w (K,N,4,4) -> wp[tap][K][N]
__global__ void pack_w_kernel(const float* __restrict__ w, _Float16* __restrict__ wp,
                              int K, int N) {
  size_t idx = (size_t)blockIdx.x * 256 + threadIdx.x;
  size_t total = 16ull * K * N;
  if (idx >= total) return;
  int co = (int)(idx % N);
  size_t r = idx / N;
  int ci  = (int)(r % K);
  int tap = (int)(r / K);
  wp[idx] = (_Float16)w[((size_t)ci * N + co) * 16 + tap];
}

// ============================================================
// 3) WMMA fragment helpers (ISA 7.12.2 layouts, f16 16x16x32)
// ============================================================
__device__ __forceinline__ half16 frag_a(const _Float16* __restrict__ base, int m,
                                         int ldk, int k0, int lane) {
  // A 16x32 f16: lanes 0-15 -> M rows, group g=lane>>4 selects K halves
  const _Float16* p = base + (size_t)(m + (lane & 15)) * ldk + k0 + ((lane >> 4) << 3);
  half8 lo = *(const half8*)(p);
  half8 hi = *(const half8*)(p + 16);
  return __builtin_shufflevector(lo, hi, 0, 1, 2, 3, 4, 5, 6, 7, 8, 9, 10, 11, 12, 13, 14, 15);
}
__device__ __forceinline__ half16 frag_b(const _Float16* __restrict__ base, int k0,
                                         int ldn, int n0, int lane) {
  // B 32x16 f16: lane -> N column, group g=lane>>4 selects K halves (same k order as A)
  const _Float16* p = base + (size_t)(k0 + ((lane >> 4) << 3)) * ldn + n0 + (lane & 15);
  half16 v;
#pragma unroll
  for (int s = 0; s < 8; ++s) {
    v[s] = p[(size_t)s * ldn];
    v[s + 8] = p[(size_t)(s + 16) * ldn];
  }
  return v;
}

// ============================================================
// 4) unified ConvTranspose-as-GEMM WMMA kernel
//    mode 0: 1x1 input (layer1), weight slice = output pixel
//    mode 1: k=4, s=2, p=1 ConvT, <=4 parity taps per output pixel
//    A: [Pin][M][K] f16, W: [16][K][N] f16, Out: [Pout][M][N] f16
//    block = 256 thr = 8 waves -> 128(M) x 128(N) tile, 32x64 per wave
// ============================================================
__global__ __launch_bounds__(256) void convt_wmma_kernel(
    const _Float16* __restrict__ A, const _Float16* __restrict__ W,
    _Float16* __restrict__ Out, int M, int K, int N,
    int Hin, int Win, int Wout, int mode) {
  const int opix = blockIdx.x;
  int nt = 0, ipix[4], wsl[4];
  if (mode == 0) {
    nt = 1; ipix[0] = 0; wsl[0] = opix;
  } else {
    const int oy = opix / Wout, ox = opix % Wout;
    int tys[2], iys[2], nty = 0;
    int yy = oy >> 1;
    if ((oy & 1) == 0) {                 // o even -> t odd: t=1 (i=yy), t=3 (i=yy-1)
      tys[nty] = 1; iys[nty] = yy; ++nty;
      if (yy - 1 >= 0) { tys[nty] = 3; iys[nty] = yy - 1; ++nty; }
    } else {                             // o odd -> t even: t=0 (i=yy+1), t=2 (i=yy)
      if (yy + 1 < Hin) { tys[nty] = 0; iys[nty] = yy + 1; ++nty; }
      tys[nty] = 2; iys[nty] = yy; ++nty;
    }
    int txs[2], ixs[2], ntx = 0;
    int xx = ox >> 1;
    if ((ox & 1) == 0) {
      txs[ntx] = 1; ixs[ntx] = xx; ++ntx;
      if (xx - 1 >= 0) { txs[ntx] = 3; ixs[ntx] = xx - 1; ++ntx; }
    } else {
      if (xx + 1 < Win) { txs[ntx] = 0; ixs[ntx] = xx + 1; ++ntx; }
      txs[ntx] = 2; ixs[ntx] = xx; ++ntx;
    }
    for (int a = 0; a < nty; ++a)
      for (int c = 0; c < ntx; ++c) {
        ipix[nt] = iys[a] * Win + ixs[c];
        wsl[nt]  = tys[a] * 4 + txs[c];
        ++nt;
      }
  }

  const int tid = threadIdx.x, lane = tid & 31, wave = tid >> 5;
  const int m0 = blockIdx.y * 128 + (wave >> 1) * 32;   // 4 waves along M
  const int n0 = blockIdx.z * 128 + (wave & 1) * 64;    // 2 waves along N

  f32x8 acc[2][4];
#pragma unroll
  for (int i = 0; i < 2; ++i)
#pragma unroll
    for (int j = 0; j < 4; ++j) { f32x8 zr = {}; acc[i][j] = zr; }

  for (int tnum = 0; tnum < nt; ++tnum) {
    const _Float16* __restrict__ Ab = A + (size_t)ipix[tnum] * M * K;
    const _Float16* __restrict__ Wb = W + (size_t)wsl[tnum] * K * N;
    for (int k0 = 0; k0 < K; k0 += 32) {
      __builtin_prefetch(Ab + (size_t)(m0 + (lane & 15)) * K + k0 + 32, 0, 0);
      half16 af[2], bf[4];
#pragma unroll
      for (int i = 0; i < 2; ++i) af[i] = frag_a(Ab, m0 + 16 * i, K, k0, lane);
#pragma unroll
      for (int j = 0; j < 4; ++j) bf[j] = frag_b(Wb, k0, N, n0 + 16 * j, lane);
#pragma unroll
      for (int i = 0; i < 2; ++i)
#pragma unroll
        for (int j = 0; j < 4; ++j)
          acc[i][j] = __builtin_amdgcn_wmma_f32_16x16x32_f16(
              false, af[i], false, bf[j], (short)0, acc[i][j], false, false);
    }
  }

  _Float16* __restrict__ Ob = Out + (size_t)opix * M * N;
  const int g = lane >> 4, ln = lane & 15;
#pragma unroll
  for (int i = 0; i < 2; ++i)
#pragma unroll
    for (int j = 0; j < 4; ++j)
#pragma unroll
      for (int r = 0; r < 8; ++r)
        Ob[(size_t)(m0 + 16 * i + r + 8 * g) * N + (n0 + 16 * j + ln)] =
            (_Float16)acc[i][j][r];
}

// ============================================================
// 5) BatchNorm statistics: one block per channel (layout [..][C])
// ============================================================
__global__ __launch_bounds__(256) void bn_stats_kernel(const _Float16* __restrict__ raw,
                                                       float* __restrict__ mean,
                                                       float* __restrict__ invstd,
                                                       int C, size_t PM) {
  __shared__ float s1[256], s2[256];
  const int c = blockIdx.x, t = threadIdx.x;
  float a = 0.0f, b = 0.0f;
  for (size_t e = t; e < PM; e += 256) {
    float v = (float)raw[e * (size_t)C + c];
    a += v; b += v * v;
  }
  s1[t] = a; s2[t] = b; __syncthreads();
  for (int s = 128; s > 0; s >>= 1) {
    if (t < s) { s1[t] += s1[t + s]; s2[t] += s2[t + s]; }
    __syncthreads();
  }
  if (t == 0) {
    float m = s1[0] / (float)PM;
    float var = s2[0] / (float)PM - m * m;
    mean[c] = m;
    invstd[c] = rsqrtf(var + BN_EPS);
  }
}

// 6) normalize + gamma/beta + tanh -> f16 activation
__global__ void bn_tanh_kernel(const _Float16* __restrict__ raw, _Float16* __restrict__ act,
                               const float* __restrict__ mean, const float* __restrict__ invstd,
                               const float* __restrict__ gamma, const float* __restrict__ beta,
                               int C, size_t total) {
  size_t idx = (size_t)blockIdx.x * 256 + threadIdx.x;
  if (idx >= total) return;
  int c = (int)(idx % C);
  float v = (float)raw[idx];
  float xn = (v - mean[c]) * invstd[c];
  act[idx] = (_Float16)tanhf(xn * gamma[c] + beta[c]);
}

// ============================================================
// 7) last ConvT (Cout=1) + sigmoid: GEMV-style, one thread/output
//    act3: [256(pix)][B][128], w4: [128][16], out: [B][32][32] f32
// ============================================================
__global__ void layer4_kernel(const _Float16* __restrict__ act3,
                              const float* __restrict__ w4,
                              const float* __restrict__ b4,
                              float* __restrict__ out) {
  size_t idx = (size_t)blockIdx.x * 256 + threadIdx.x;
  if (idx >= (size_t)BATCH * 32 * 32) return;
  const int ox = (int)(idx % 32);
  const int oy = (int)((idx / 32) % 32);
  const int b  = (int)(idx / 1024);

  int tys[2], iys[2], nty = 0;
  int yy = oy >> 1;
  if ((oy & 1) == 0) {
    tys[nty] = 1; iys[nty] = yy; ++nty;
    if (yy - 1 >= 0) { tys[nty] = 3; iys[nty] = yy - 1; ++nty; }
  } else {
    if (yy + 1 < 16) { tys[nty] = 0; iys[nty] = yy + 1; ++nty; }
    tys[nty] = 2; iys[nty] = yy; ++nty;
  }
  int txs[2], ixs[2], ntx = 0;
  int xx = ox >> 1;
  if ((ox & 1) == 0) {
    txs[ntx] = 1; ixs[ntx] = xx; ++ntx;
    if (xx - 1 >= 0) { txs[ntx] = 3; ixs[ntx] = xx - 1; ++ntx; }
  } else {
    if (xx + 1 < 16) { txs[ntx] = 0; ixs[ntx] = xx + 1; ++ntx; }
    txs[ntx] = 2; ixs[ntx] = xx; ++ntx;
  }

  float acc = b4[0];
  for (int a = 0; a < nty; ++a)
    for (int c = 0; c < ntx; ++c) {
      const _Float16* ap = act3 + ((size_t)(iys[a] * 16 + ixs[c]) * BATCH + b) * 128;
      const int tap = tys[a] * 4 + txs[c];
      float s = 0.0f;
#pragma unroll 8
      for (int ci = 0; ci < 128; ++ci) s += (float)ap[ci] * w4[ci * 16 + tap];
      acc += s;
    }
  out[idx] = 1.0f / (1.0f + expf(-acc));
}

// ============================================================
// launcher
// ============================================================
extern "C" void kernel_launch(void* const* d_in, const int* in_sizes, int n_in,
                              void* d_out, int out_size, void* d_ws, size_t ws_size,
                              hipStream_t stream) {
  const float* x   = (const float*)d_in[0];
  const float* w1  = (const float*)d_in[1];
  const float* g1  = (const float*)d_in[3];
  const float* be1 = (const float*)d_in[4];
  const float* w2  = (const float*)d_in[5];
  const float* g2  = (const float*)d_in[7];
  const float* be2 = (const float*)d_in[8];
  const float* w3  = (const float*)d_in[9];
  const float* g3  = (const float*)d_in[11];
  const float* be3 = (const float*)d_in[12];
  const float* w4  = (const float*)d_in[13];
  const float* b4  = (const float*)d_in[14];
  float* out = (float*)d_out;

  char* ws = (char*)d_ws;
  size_t off = 0;
  auto take = [&](size_t bytes) { char* p = ws + off; off += (bytes + 255) & ~(size_t)255; return p; };
  _Float16* Z   = (_Float16*)take((size_t)BATCH * NZP * 2);              // 4 MB
  _Float16* W1P = (_Float16*)take(16ull * NZP * 512 * 2);                // 16.8 MB
  _Float16* W2P = (_Float16*)take(16ull * 512 * 256 * 2);                // 4.2 MB
  _Float16* W3P = (_Float16*)take(16ull * 256 * 128 * 2);                // 1 MB
  float*    MEAN = (float*)take(512 * 4);
  float*    ISTD = (float*)take(512 * 4);
  const size_t REG = 256ull * BATCH * 128 * 2;                           // 134 MB
  _Float16* R0 = (_Float16*)take(REG);
  _Float16* R1 = (_Float16*)take(REG);
  (void)ws_size; (void)in_sizes; (void)n_in; (void)out_size;

  // --- z = truncation_vector(x) ---
  topk_kernel<<<BATCH, 256, 0, stream>>>(x, Z);

  // --- pack weights to f16 [slice][K][N] ---
  {
    size_t t1 = 16ull * NZP * 512;
    pack_w1_kernel<<<(unsigned)((t1 + 255) / 256), 256, 0, stream>>>(w1, W1P);
    size_t t2 = 16ull * 512 * 256;
    pack_w_kernel<<<(unsigned)((t2 + 255) / 256), 256, 0, stream>>>(w2, W2P, 512, 256);
    size_t t3 = 16ull * 256 * 128;
    pack_w_kernel<<<(unsigned)((t3 + 255) / 256), 256, 0, stream>>>(w3, W3P, 256, 128);
  }

  // --- layer 1: GEMM per output pixel (16), K=1024, N=512 -> raw1 in R0 ---
  convt_wmma_kernel<<<dim3(16, BATCH / 128, 512 / 128), 256, 0, stream>>>(
      Z, W1P, R0, BATCH, NZP, 512, 1, 1, 4, 0);
  bn_stats_kernel<<<512, 256, 0, stream>>>(R0, MEAN, ISTD, 512, 16ull * BATCH);
  {
    size_t tot = 16ull * BATCH * 512;
    bn_tanh_kernel<<<(unsigned)((tot + 255) / 256), 256, 0, stream>>>(
        R0, R1, MEAN, ISTD, g1, be1, 512, tot);                          // act1 -> R1
  }

  // --- layer 2: ConvT 512->256, 4x4 -> 8x8 ---
  convt_wmma_kernel<<<dim3(64, BATCH / 128, 256 / 128), 256, 0, stream>>>(
      R1, W2P, R0, BATCH, 512, 256, 4, 4, 8, 1);                         // raw2 -> R0
  bn_stats_kernel<<<256, 256, 0, stream>>>(R0, MEAN, ISTD, 256, 64ull * BATCH);
  {
    size_t tot = 64ull * BATCH * 256;
    bn_tanh_kernel<<<(unsigned)((tot + 255) / 256), 256, 0, stream>>>(
        R0, R1, MEAN, ISTD, g2, be2, 256, tot);                          // act2 -> R1
  }

  // --- layer 3: ConvT 256->128, 8x8 -> 16x16 ---
  convt_wmma_kernel<<<dim3(256, BATCH / 128, 128 / 128), 256, 0, stream>>>(
      R1, W3P, R0, BATCH, 256, 128, 8, 8, 16, 1);                        // raw3 -> R0
  bn_stats_kernel<<<128, 256, 0, stream>>>(R0, MEAN, ISTD, 128, 256ull * BATCH);
  {
    size_t tot = 256ull * BATCH * 128;
    bn_tanh_kernel<<<(unsigned)((tot + 255) / 256), 256, 0, stream>>>(
        R0, R1, MEAN, ISTD, g3, be3, 128, tot);                          // act3 -> R1
  }

  // --- layer 4: ConvT 128->1 + sigmoid ---
  {
    size_t tot = (size_t)BATCH * 32 * 32;
    layer4_kernel<<<(unsigned)((tot + 255) / 256), 256, 0, stream>>>(R1, w4, b4, out);
  }
}
